// SNN_18227841204760
// MI455X (gfx1250) — matrix operand order repaired and checked by reading, not compiled, not run on hardware
//
#include <hip/hip_runtime.h>

typedef _Float16 v16h __attribute__((ext_vector_type(16)));
typedef float    v8f  __attribute__((ext_vector_type(8)));
typedef __fp16   h2   __attribute__((ext_vector_type(2)));   // cvt_pkrtz result type

#define BB 512
#define TT 100
#define DD 784
#define HH 128
#define OO 10
#define BT (BB*TT)      // 51200 GEMM rows
#define SA 40           // LDS row stride in f16 elements (80B, conflict-free)
#define NCH 25          // K chunks: 24 full x32 + one 16-wide tail

union Frag16 { v16h v; uint4 u[2]; };
union Pack16 { h2 h[8]; uint4 u[2]; };

// ---------------------------------------------------------------------------
// Kernel 1: cur1[m][h] = sum_d x[m][d] * W1[h][d] + b1[h]
// m = b*T + t (x's natural row order), stored to workspace as [B*T][H] fp32.
// WMMA f32_16x16x32_f16, block tile 128x128, ping-pong LDS, one barrier per
// K-chunk, global loads for chunk i+1 overlapped with WMMA on chunk i.
// ---------------------------------------------------------------------------
__global__ __launch_bounds__(256) void snn_fc1_wmma(
    const float* __restrict__ x, const float* __restrict__ W1,
    const float* __restrict__ b1, float* __restrict__ cur1)
{
    __shared__ _Float16 As[2][128 * SA];  // x tile   (rows m_local, cols k)
    __shared__ _Float16 Ws[2][128 * SA];  // W1 tile  (rows h,       cols k)

    const int tid   = threadIdx.x;
    const int wave  = tid >> 5;
    const int lane  = tid & 31;
    const int lidx  = lane & 15;
    const int lhalf = lane >> 4;
    const int mblk  = blockIdx.x * 128;

    // staging: 2 threads per row, 16 columns each
    const int srow = tid >> 1;
    const int c0   = (tid & 1) * 16;
    const float* xrow = x  + (size_t)(mblk + srow) * DD + c0;
    const float* wrow = W1 + (size_t)srow * DD + c0;

    // accumulators: 8 N-tiles, bias-initialized (bias is per-column h)
    v8f acc[8];
#pragma unroll
    for (int nt = 0; nt < 8; ++nt) {
        const float bias = b1[nt * 16 + lidx];
#pragma unroll
        for (int r = 0; r < 8; ++r) acc[nt][r] = bias;
    }

    // fragment addressing (ISA 7.12.2 layouts)
    const int arow = 16 * wave + lidx;   // A: lane holds row (M)
    const int k0a  = lhalf * 8;          // A: lanes>=16 start at K=8 (and +16)
    const int k0b  = lhalf * 8;          // B: lanes>=16 start at K=8 (and +16)

    // ---- prologue: load chunk 0 (always full) into registers ----
    float4 xv[4], wv[4];
    {
        const float4* xp = (const float4*)xrow;
        const float4* wp = (const float4*)wrow;
        xv[0] = xp[0]; xv[1] = xp[1]; xv[2] = xp[2]; xv[3] = xp[3];
        wv[0] = wp[0]; wv[1] = wp[1]; wv[2] = wp[2]; wv[3] = wp[3];
        __builtin_prefetch(xrow + 32);   // global_prefetch_b8
        __builtin_prefetch(wrow + 32);
    }

    for (int i = 0; i < NCH; ++i) {
        const int p = i & 1;

        // ---- convert chunk i to f16 and store to LDS buffer p ----
        Pack16 px, pw;
#pragma unroll
        for (int j = 0; j < 4; ++j) {
            px.h[2*j]   = __builtin_amdgcn_cvt_pkrtz(xv[j].x, xv[j].y);
            px.h[2*j+1] = __builtin_amdgcn_cvt_pkrtz(xv[j].z, xv[j].w);
            pw.h[2*j]   = __builtin_amdgcn_cvt_pkrtz(wv[j].x, wv[j].y);
            pw.h[2*j+1] = __builtin_amdgcn_cvt_pkrtz(wv[j].z, wv[j].w);
        }
        uint4* ad = (uint4*)&As[p][srow * SA + c0];
        uint4* wd = (uint4*)&Ws[p][srow * SA + c0];
        ad[0] = px.u[0];  ad[1] = px.u[1];
        wd[0] = pw.u[0];  wd[1] = pw.u[1];
        __syncthreads();   // single barrier per chunk (ping-pong makes it safe)

        // ---- issue global loads for chunk i+1; they overlap the WMMAs ----
        if (i + 1 < NCH) {
            const int kb = 32 * (i + 1);
            if ((kb + c0) < DD) {        // false only for tail's upper half
                const float4* xp = (const float4*)(xrow + kb);
                const float4* wp = (const float4*)(wrow + kb);
                xv[0] = xp[0]; xv[1] = xp[1]; xv[2] = xp[2]; xv[3] = xp[3];
                wv[0] = wp[0]; wv[1] = wp[1]; wv[2] = wp[2]; wv[3] = wp[3];
                if (kb + 32 < DD) {
                    __builtin_prefetch(xrow + kb + 32);
                    __builtin_prefetch(wrow + kb + 32);
                }
            } else {
#pragma unroll
                for (int j = 0; j < 4; ++j) {
                    xv[j].x = xv[j].y = xv[j].z = xv[j].w = 0.f;
                    wv[j].x = wv[j].y = wv[j].z = wv[j].w = 0.f;
                }
            }
        }

        // ---- A fragment: row = arow, K = {k0a..+7} ++ {k0a+16..+23} ----
        Frag16 af;
        const _Float16* abase = &As[p][arow * SA];
        af.u[0] = *(const uint4*)(abase + k0a);
        af.u[1] = *(const uint4*)(abase + k0a + 16);

        // ---- 8 B fragments, software-pipelined (bf double buffer) ----
        const _Float16* wsb = &Ws[p][lidx * SA + k0b];
        Frag16 bf[2];
        bf[0].u[0] = ((const uint4*)wsb)[0];
        bf[0].u[1] = ((const uint4*)wsb)[2];
#pragma unroll
        for (int nt = 0; nt < 8; ++nt) {
            if (nt < 7) {
                const _Float16* bb = wsb + (nt + 1) * 16 * SA;
                bf[(nt + 1) & 1].u[0] = ((const uint4*)bb)[0];
                bf[(nt + 1) & 1].u[1] = ((const uint4*)bb)[2];
            }
            acc[nt] = __builtin_amdgcn_wmma_f32_16x16x32_f16(
                false, af.v, false, bf[nt & 1].v, (short)0, acc[nt], false, false);
        }
        // no trailing barrier: next iteration writes the *other* buffer, and
        // the top-of-loop barrier already orders reads(i) vs writes(i+2).
    }

    // ---- store: C/D layout: VGPR r -> row (lhalf*8 + r), col = lane&15 ----
    const int mbase = mblk + 16 * wave + lhalf * 8;
#pragma unroll
    for (int nt = 0; nt < 8; ++nt) {
        const int col = nt * 16 + lidx;
#pragma unroll
        for (int r = 0; r < 8; ++r) {
            cur1[(size_t)(mbase + r) * HH + col] = acc[nt][r];
        }
    }
}

// ---------------------------------------------------------------------------
// Kernel 2: sequential LIF scan over T. One wave32 per batch element b.
// Lane holds mem1 for h = 4*lane..4*lane+3; lanes 0..9 carry mem2.
// ---------------------------------------------------------------------------
__global__ __launch_bounds__(256) void snn_scan(
    const float* __restrict__ cur1, const float* __restrict__ W2,
    const float* __restrict__ b2, float* __restrict__ out)
{
    const int lane = threadIdx.x & 31;
    const int b    = blockIdx.x * 8 + (threadIdx.x >> 5);

    float4 w2r[OO];
#pragma unroll
    for (int o = 0; o < OO; ++o)
        w2r[o] = *(const float4*)(W2 + o * HH + lane * 4);
    float b2r[OO];
#pragma unroll
    for (int o = 0; o < OO; ++o) b2r[o] = b2[o];

    float4 m1; m1.x = m1.y = m1.z = m1.w = 0.f;
    float  m2 = 0.f;
    const size_t TB10 = (size_t)TT * BB * OO;

    for (int t = 0; t < TT; ++t) {
        float4 c = *(const float4*)(cur1 + ((size_t)b * TT + t) * HH + lane * 4);

        // LIF layer 1 (reset-by-subtract, beta=0.95, thr=1)
        float rs, s0, s1, s2, s3;
        rs = m1.x > 1.f ? 1.f : 0.f; m1.x = 0.95f * m1.x + c.x - rs; s0 = m1.x > 1.f ? 1.f : 0.f;
        rs = m1.y > 1.f ? 1.f : 0.f; m1.y = 0.95f * m1.y + c.y - rs; s1 = m1.y > 1.f ? 1.f : 0.f;
        rs = m1.z > 1.f ? 1.f : 0.f; m1.z = 0.95f * m1.z + c.z - rs; s2 = m1.z > 1.f ? 1.f : 0.f;
        rs = m1.w > 1.f ? 1.f : 0.f; m1.w = 0.95f * m1.w + c.w - rs; s3 = m1.w > 1.f ? 1.f : 0.f;

        // cur2[o] = spk1 . W2[o][:] + b2[o]  (wave reduction via ds shuffles)
        float tot[OO];
#pragma unroll
        for (int o = 0; o < OO; ++o) {
            float p = s0 * w2r[o].x;
            p = fmaf(s1, w2r[o].y, p);
            p = fmaf(s2, w2r[o].z, p);
            p = fmaf(s3, w2r[o].w, p);
#pragma unroll
            for (int off = 16; off > 0; off >>= 1)
                p += __shfl_xor(p, off, 32);
            tot[o] = p + b2r[o];
        }

        // lane o picks its own cur2
        float c2 = tot[0];
#pragma unroll
        for (int o = 1; o < OO; ++o) c2 = (lane == o) ? tot[o] : c2;

        if (lane < OO) {
            const float rs2 = m2 > 1.f ? 1.f : 0.f;
            m2 = 0.95f * m2 + c2 - rs2;
            const float spk2 = m2 > 1.f ? 1.f : 0.f;
            const size_t oidx = ((size_t)t * BB + b) * OO + lane;
            out[oidx]        = spk2;  // spk2_rec block
            out[TB10 + oidx] = m2;    // mem2_rec block
        }
    }
}

// ---------------------------------------------------------------------------
extern "C" void kernel_launch(void* const* d_in, const int* in_sizes, int n_in,
                              void* d_out, int out_size, void* d_ws, size_t ws_size,
                              hipStream_t stream) {
    const float* x  = (const float*)d_in[0];
    const float* W1 = (const float*)d_in[1];
    const float* b1 = (const float*)d_in[2];
    const float* W2 = (const float*)d_in[3];
    const float* b2 = (const float*)d_in[4];
    float* out  = (float*)d_out;
    float* cur1 = (float*)d_ws;   // [B*T][H] fp32 = 26.2 MB scratch

    snn_fc1_wmma<<<dim3(BT / 128), dim3(256), 0, stream>>>(x, W1, b1, cur1);
    snn_scan   <<<dim3(BB / 8),   dim3(256), 0, stream>>>(cur1, W2, b2, out);
}
